// FCBlock_89936615178932
// MI455X (gfx1250) — matrix operand order repaired and checked by reading, not compiled, not run on hardware
//
#include <hip/hip_runtime.h>

// ---------------------------------------------------------------------------
// Transformer block (LN -> MHA(causal) -> +res -> LN -> sigmoid-FFN -> +res)
// for MI455X (gfx1250, wave32). All matmuls via v_wmma_f32_16x16x32_f16.
// ---------------------------------------------------------------------------

typedef __attribute__((ext_vector_type(16))) _Float16 v16h;
typedef __attribute__((ext_vector_type(8)))  _Float16 h8;
typedef __attribute__((ext_vector_type(8)))  float    v8f;

#define NB   4
#define NT   2048
#define NC   768
#define NH   12
#define HD   64
#define NF   1536
#define NM   (NB * NT)   // 8192 rows

// 16-lane (half-wave) max butterfly via DPP16: XOR1, XOR2, row_mirror(+15),
// row_half_mirror(+7). Stays within each 16-lane group (wave32 = 2 DPP rows).
__device__ __forceinline__ float max16(float x)
{
    int v;
    v = __builtin_amdgcn_update_dpp(0, __float_as_int(x), 0xB1,  0xF, 0xF, true);
    x = fmaxf(x, __int_as_float(v));
    v = __builtin_amdgcn_update_dpp(0, __float_as_int(x), 0x4E,  0xF, 0xF, true);
    x = fmaxf(x, __int_as_float(v));
    v = __builtin_amdgcn_update_dpp(0, __float_as_int(x), 0x140, 0xF, 0xF, true);
    x = fmaxf(x, __int_as_float(v));
    v = __builtin_amdgcn_update_dpp(0, __float_as_int(x), 0x141, 0xF, 0xF, true);
    x = fmaxf(x, __int_as_float(v));
    return x;
}

// ------------------------------ LayerNorm ----------------------------------
__global__ __launch_bounds__(256)
void ln_kernel(const float* __restrict__ x, const float* __restrict__ g,
               const float* __restrict__ b, _Float16* __restrict__ out)
{
    __shared__ float s1[256], s2[256];
    const int row = blockIdx.x;
    const int t = threadIdx.x;
    const float* xr = x + (size_t)row * NC;
    float v0 = xr[t], v1 = xr[t + 256], v2 = xr[t + 512];
    s1[t] = v0 + v1 + v2;
    s2[t] = v0 * v0 + v1 * v1 + v2 * v2;
    __syncthreads();
    for (int off = 128; off > 0; off >>= 1) {
        if (t < off) { s1[t] += s1[t + off]; s2[t] += s2[t + off]; }
        __syncthreads();
    }
    const float mean = s1[0] * (1.0f / NC);
    const float var  = s2[0] * (1.0f / NC) - mean * mean;
    const float inv  = rsqrtf(var + 1e-5f);
    _Float16* orow = out + (size_t)row * NC;
    orow[t]       = (_Float16)((v0 - mean) * inv * g[t]       + b[t]);
    orow[t + 256] = (_Float16)((v1 - mean) * inv * g[t + 256] + b[t + 256]);
    orow[t + 512] = (_Float16)((v2 - mean) * inv * g[t + 512] + b[t + 512]);
}

// --------------------- weight repack (fp32 -> f16, [N,K]) ------------------
// src [H, C, hd] -> dst [(h*64+d), c]
__global__ __launch_bounds__(256)
void conv_qkv(const float* __restrict__ src, _Float16* __restrict__ dst)
{
    int tid = blockIdx.x * 256 + threadIdx.x;        // NC*NC total (exact)
    int c = tid % NC, n = tid / NC;
    int h = n >> 6, d = n & 63;
    dst[(size_t)n * NC + c] = (_Float16)src[((size_t)h * NC + c) * HD + d];
}

// src [K, N] fp32 -> dst [N, K] f16
__global__ __launch_bounds__(256)
void conv_T(const float* __restrict__ src, _Float16* __restrict__ dst,
            int K, int N)
{
    int tid = blockIdx.x * 256 + threadIdx.x;
    if (tid >= K * N) return;
    int k = tid / N, n = tid % N;
    dst[(size_t)n * K + k] = (_Float16)src[tid];
}

// ------------------------------ WMMA GEMM ----------------------------------
// Out = A[M,K](f16) * Wt[N,K](f16, row=N)  (+ bias / residual per epilogue)
// block = 256 thr = 8 waves; block tile 128(M) x 64(N); wave tile 16 x 64.
enum { EPI_H16 = 0, EPI_H16T = 1, EPI_RESID = 2, EPI_SIG = 3 };

template <int EPI>
__global__ __launch_bounds__(256)
void gemm_wmma(const _Float16* __restrict__ A, const _Float16* __restrict__ Wt,
               const float* __restrict__ bias, const float* __restrict__ resid,
               void* __restrict__ outv, int M, int N, int K)
{
    const int lane  = threadIdx.x & 31;
    const int wave  = threadIdx.x >> 5;
    const int l15   = lane & 15;
    const int hi    = (lane >> 4) & 1;            // 0: lanes 0-15, 1: 16-31
    const int nbase = blockIdx.x * 64;
    const int mbase = blockIdx.y * 128 + wave * 16;

    const _Float16* arow = A + (size_t)(mbase + l15) * K;
    v8f acc[4] = {};

#pragma unroll 2
    for (int k0 = 0; k0 < K; k0 += 32) {
        if (k0 + 32 < K) __builtin_prefetch(arow + k0 + 32, 0, 0);
        // A fragment, 16x32 f16: lane row = l15; K runs [hi*16 .. +15]
        h8 alo = *(const h8*)(arow + k0 + hi * 16);
        h8 ahi = *(const h8*)(arow + k0 + hi * 16 + 8);
        v16h a;
#pragma unroll
        for (int i = 0; i < 8; ++i) { a[i] = alo[i]; a[i + 8] = ahi[i]; }
#pragma unroll
        for (int nt = 0; nt < 4; ++nt) {
            // B fragment, 32x16: lane col = l15, K = i + 16*hi (contig in Wt[N,K])
            const _Float16* bp =
                Wt + (size_t)(nbase + nt * 16 + l15) * K + k0 + hi * 16;
            v16h bfr = *(const v16h*)bp;
            acc[nt] = __builtin_amdgcn_wmma_f32_16x16x32_f16(
                false, a, false, bfr, (short)0, acc[nt], false, false);
        }
    }

#pragma unroll
    for (int nt = 0; nt < 4; ++nt) {
        const int n = nbase + nt * 16 + l15;
#pragma unroll
        for (int r = 0; r < 8; ++r) {
            const int m = mbase + r + hi * 8;     // C layout: M = r + 8*hi
            float v = acc[nt][r];
            if (EPI == EPI_H16) {
                ((_Float16*)outv)[(size_t)m * N + n] = (_Float16)v;
            } else if (EPI == EPI_H16T) {         // transposed store (for V)
                ((_Float16*)outv)[(size_t)n * M + m] = (_Float16)v;
            } else if (EPI == EPI_RESID) {        // out = resid + A*W + bias
                ((float*)outv)[(size_t)m * N + n] =
                    resid[(size_t)m * N + n] + v + bias[n];
            } else {                              // EPI_SIG: sigmoid(A*W + b)
                float s = v + bias[n];
                ((_Float16*)outv)[(size_t)m * N + n] =
                    (_Float16)(1.0f / (1.0f + __expf(-s)));
            }
        }
    }
}

// --------------------------- flash attention -------------------------------
// Q [B*T, H*hd] f16, K [B*T, H*hd] f16, Vt [H*hd, B*T] f16 -> attn f16
// block = 8 waves; each wave owns a 16-query tile; 32-key tiles, online softmax.
// Row max:   DPP16 butterfly (no LDS round-trips).
// Row sum:   one extra WMMA against an all-ones B fragment (P @ 1 -> rowsums
//            replicated across all 16 columns, already in C-layout).
__global__ __launch_bounds__(256)
void flash_attn(const _Float16* __restrict__ Q, const _Float16* __restrict__ Km,
                const _Float16* __restrict__ Vt, _Float16* __restrict__ attn)
{
    __shared__ _Float16 Pbuf[8][16 * 32];         // per-wave P tile (16x32)
    const int lane  = threadIdx.x & 31;
    const int wave  = threadIdx.x >> 5;
    const int l15   = lane & 15;
    const int hi    = (lane >> 4) & 1;
    const int bh    = blockIdx.y;                 // 0..B*H-1
    const int b     = bh / NH, h = bh % NH;
    const int qbase = blockIdx.x * 128 + wave * 16;

    // all-ones B fragment for WMMA row-sum
    v16h vones;
#pragma unroll
    for (int i = 0; i < 16; ++i) vones[i] = (_Float16)1.0f;

    // preload Q A-fragments (hd=64 -> 2 K-steps of 32)
    const _Float16* qrow = Q + ((size_t)(b * NT + qbase + l15)) * NC + h * HD;
    v16h qa[2];
#pragma unroll
    for (int ks = 0; ks < 2; ++ks) {
        h8 lo = *(const h8*)(qrow + ks * 32 + hi * 16);
        h8 hp = *(const h8*)(qrow + ks * 32 + hi * 16 + 8);
#pragma unroll
        for (int i = 0; i < 8; ++i) { qa[ks][i] = lo[i]; qa[ks][i + 8] = hp[i]; }
    }

    v8f o[4] = {};
    float rmax[8], rsum[8];
#pragma unroll
    for (int r = 0; r < 8; ++r) { rmax[r] = -3.0e38f; rsum[r] = 0.0f; }

    const int ks_last = (qbase + 15) >> 5;
    for (int ks = 0; ks <= ks_last; ++ks) {
        const int sbase = ks * 32;
        // scores S[16q x 32k] = Q Kt : two 16x16 C-tiles, 2 K-steps each
        v8f s0 = {}, s1 = {};
#pragma unroll
        for (int kk = 0; kk < 2; ++kk) {
            v16h b0 = *(const v16h*)(Km + ((size_t)(b * NT + sbase + l15)) * NC
                                     + h * HD + kk * 32 + hi * 16);
            v16h b1 = *(const v16h*)(Km + ((size_t)(b * NT + sbase + 16 + l15)) * NC
                                     + h * HD + kk * 32 + hi * 16);
            s0 = __builtin_amdgcn_wmma_f32_16x16x32_f16(
                false, qa[kk], false, b0, (short)0, s0, false, false);
            s1 = __builtin_amdgcn_wmma_f32_16x16x32_f16(
                false, qa[kk], false, b1, (short)0, s1, false, false);
        }
        const float scale = 0.125f;               // 1/sqrt(64)
        const bool do_mask = (ks == ks_last);
        float corr[8];
#pragma unroll
        for (int r = 0; r < 8; ++r) {
            float a0 = s0[r] * scale, a1 = s1[r] * scale;
            if (do_mask) {
                int qg = qbase + r + hi * 8;
                if (sbase + l15 > qg)      a0 = -3.0e38f;
                if (sbase + 16 + l15 > qg) a1 = -3.0e38f;
            }
            // row max across the 16 lanes holding this row's columns
            const float m  = max16(fmaxf(a0, a1));
            const float nm = fmaxf(rmax[r], m);
            corr[r] = __expf(rmax[r] - nm);
            rmax[r] = nm;
            const float p0 = __expf(a0 - nm), p1 = __expf(a1 - nm);
            // stash P in C-layout -> LDS (row-major 16x32)
            Pbuf[wave][(r + hi * 8) * 32 + l15]      = (_Float16)p0;
            Pbuf[wave][(r + hi * 8) * 32 + 16 + l15] = (_Float16)p1;
#pragma unroll
            for (int nt = 0; nt < 4; ++nt) o[nt][r] *= corr[r];
        }
        asm volatile("s_wait_dscnt 0" ::: "memory");
        // reload P as 16x32 A-fragment from LDS
        const _Float16* pb = &Pbuf[wave][0];
        h8 plo = *(const h8*)(pb + l15 * 32 + hi * 16);
        h8 phi = *(const h8*)(pb + l15 * 32 + hi * 16 + 8);
        v16h pa;
#pragma unroll
        for (int i = 0; i < 8; ++i) { pa[i] = plo[i]; pa[i + 8] = phi[i]; }
        // row sums via WMMA: P @ ones -> every column holds the row sum
        v8f ps = {};
        ps = __builtin_amdgcn_wmma_f32_16x16x32_f16(
            false, pa, false, vones, (short)0, ps, false, false);
#pragma unroll
        for (int r = 0; r < 8; ++r) rsum[r] = rsum[r] * corr[r] + ps[r];
        // O += P @ V  (V transposed: contiguous along keys)
#pragma unroll
        for (int nt = 0; nt < 4; ++nt) {
            const _Float16* vp = Vt + ((size_t)(h * HD + nt * 16 + l15)) * NM
                                 + b * NT + sbase + hi * 16;
            v16h vb = *(const v16h*)vp;
            o[nt] = __builtin_amdgcn_wmma_f32_16x16x32_f16(
                false, pa, false, vb, (short)0, o[nt], false, false);
        }
    }
    // finalize: divide by row sums, store f16 [B*T, H*hd]
#pragma unroll
    for (int r = 0; r < 8; ++r) {
        const float inv = 1.0f / rsum[r];
        const int m = qbase + r + hi * 8;
#pragma unroll
        for (int nt = 0; nt < 4; ++nt) {
            attn[((size_t)(b * NT + m)) * NC + h * HD + nt * 16 + l15] =
                (_Float16)(o[nt][r] * inv);
        }
    }
}

// ------------------------------- launcher ----------------------------------
extern "C" void kernel_launch(void* const* d_in, const int* in_sizes, int n_in,
                              void* d_out, int out_size, void* d_ws, size_t ws_size,
                              hipStream_t stream)
{
    const float* x   = (const float*)d_in[0];
    const float* Wq  = (const float*)d_in[1];
    const float* Wk  = (const float*)d_in[2];
    const float* Wv  = (const float*)d_in[3];
    const float* Wo  = (const float*)d_in[4];
    const float* bo  = (const float*)d_in[5];
    const float* W1  = (const float*)d_in[6];
    const float* b1  = (const float*)d_in[7];
    const float* W2  = (const float*)d_in[8];
    const float* b2  = (const float*)d_in[9];
    const float* g1  = (const float*)d_in[10];
    const float* be1 = (const float*)d_in[11];
    const float* g2  = (const float*)d_in[12];
    const float* be2 = (const float*)d_in[13];
    float* out = (float*)d_out;

    char* w = (char*)d_ws;
    size_t off = 0;
    auto alloc = [&](size_t bytes) -> char* {
        char* p = w + off;
        off += bytes;
        off = (off + 255) & ~(size_t)255;
        return p;
    };
    _Float16* h1   = (_Float16*)alloc((size_t)NM * NC * 2);   // ln1(x); reused as h2
    _Float16* Wqt  = (_Float16*)alloc((size_t)NC * NC * 2);
    _Float16* Wkt  = (_Float16*)alloc((size_t)NC * NC * 2);
    _Float16* Wvt  = (_Float16*)alloc((size_t)NC * NC * 2);
    _Float16* Wot  = (_Float16*)alloc((size_t)NC * NC * 2);
    _Float16* W1t  = (_Float16*)alloc((size_t)NC * NF * 2);
    _Float16* W2t  = (_Float16*)alloc((size_t)NF * NC * 2);
    _Float16* Qb   = (_Float16*)alloc((size_t)NM * NC * 2);
    _Float16* Kb   = (_Float16*)alloc((size_t)NM * NC * 2);
    _Float16* Vtb  = (_Float16*)alloc((size_t)NC * NM * 2);   // [H*hd, B*T]
    _Float16* attn = (_Float16*)alloc((size_t)NM * NC * 2);
    float*    x1   = (float*)   alloc((size_t)NM * NC * 4);
    _Float16* ffn1 = (_Float16*)alloc((size_t)NM * NF * 2);
    (void)ws_size; (void)in_sizes; (void)n_in; (void)out_size;

    // 1. h1 = ln1(x) in f16
    ln_kernel<<<NM, 256, 0, stream>>>(x, g1, be1, h1);
    // 2. repack weights to f16 [N,K]
    conv_qkv<<<(NC * NC) / 256, 256, 0, stream>>>(Wq, Wqt);
    conv_qkv<<<(NC * NC) / 256, 256, 0, stream>>>(Wk, Wkt);
    conv_qkv<<<(NC * NC) / 256, 256, 0, stream>>>(Wv, Wvt);
    conv_T<<<(NC * NC + 255) / 256, 256, 0, stream>>>(Wo, Wot, NC, NC);
    conv_T<<<(NC * NF + 255) / 256, 256, 0, stream>>>(W1, W1t, NC, NF);
    conv_T<<<(NF * NC + 255) / 256, 256, 0, stream>>>(W2, W2t, NF, NC);
    // 3. Q, K, V projections (V stored transposed for the P@V B-fragment)
    dim3 gp(NC / 64, NM / 128);
    gemm_wmma<EPI_H16 ><<<gp, 256, 0, stream>>>(h1, Wqt, nullptr, nullptr, Qb,  NM, NC, NC);
    gemm_wmma<EPI_H16 ><<<gp, 256, 0, stream>>>(h1, Wkt, nullptr, nullptr, Kb,  NM, NC, NC);
    gemm_wmma<EPI_H16T><<<gp, 256, 0, stream>>>(h1, Wvt, nullptr, nullptr, Vtb, NM, NC, NC);
    // 4. causal flash attention
    flash_attn<<<dim3(NT / 128, NB * NH), 256, 0, stream>>>(Qb, Kb, Vtb, attn);
    // 5. x1 = x + attn @ Wo + bo
    gemm_wmma<EPI_RESID><<<gp, 256, 0, stream>>>(attn, Wot, bo, x, x1, NM, NC, NC);
    // 6. h2 = ln2(x1) (reuse h1 buffer)
    ln_kernel<<<NM, 256, 0, stream>>>(x1, g2, be2, h1);
    // 7. ffn1 = sigmoid(h2 @ W1 + b1)
    gemm_wmma<EPI_SIG><<<dim3(NF / 64, NM / 128), 256, 0, stream>>>(
        h1, W1t, b1, nullptr, ffn1, NM, NF, NC);
    // 8. out = x1 + ffn1 @ W2 + b2
    gemm_wmma<EPI_RESID><<<gp, 256, 0, stream>>>(ffn1, W2t, b2, x1, out, NM, NC, NF);
}
